// AssignLayer_18794776887999
// MI455X (gfx1250) — matrix operand order repaired and checked by reading, not compiled, not run on hardware
//
#include <hip/hip_runtime.h>
#include <hip/hip_bf16.h>

typedef __attribute__((ext_vector_type(16))) _Float16 v16h;
typedef __attribute__((ext_vector_type(8)))  _Float16 v8h;
typedef __attribute__((ext_vector_type(8)))  float    v8f;
typedef __attribute__((ext_vector_type(4)))  float    v4f;

#define NPTS   1000000
#define DIMS   16
#define NJ     32
#define NL     8
#define NTILES (NPTS / 16)
#define LOG2E  1.4426950408889634f

#if __has_builtin(__builtin_amdgcn_exp2f)
#define EXP2F __builtin_amdgcn_exp2f
#else
#define EXP2F exp2f
#endif
#if __has_builtin(__builtin_amdgcn_rcpf)
#define RCPF __builtin_amdgcn_rcpf
#else
#define RCPF(x) (1.0f / (x))
#endif

// ---------------------------------------------------------------------------
// Prep kernel: repack mu / w into exact WMMA B-fragment order (f16), and
// precompute ||mu_j||^2 and a_j = -log2(e)/|s_j| so the gaussian is a single
// v_exp_f32 per (point, j).
//
//  Bmu[jt][r][h]     = mu[16*jt + r][h]          (B = mu^T; K rows 16..31 = 0)
//  Bw [t][hi][r][h]  = w[j = 16*(h>>3) + 8*hi + (h&7)][k][l],   c = 16*t + r, k=c>>3, l=c&7
// Each lane of the main kernel then loads its 16 halves (32B) contiguously.
// ---------------------------------------------------------------------------
__global__ void prep_kernel(const float* __restrict__ mu,
                            const float* __restrict__ s,
                            const float* __restrict__ w,
                            _Float16* __restrict__ Bmu,
                            _Float16* __restrict__ Bw,
                            float* __restrict__ munorm,
                            float* __restrict__ acoef) {
  const int tid = threadIdx.x;
  for (int i = tid; i < 2 * 16 * 16; i += blockDim.x) {
    int h   = i & 15;
    int row = i >> 4;                      // j = 16*jt + r
    Bmu[i] = (_Float16)mu[row * DIMS + h];
  }
  for (int i = tid; i < 8 * 2 * 16 * 16; i += blockDim.x) {
    int h  = i & 15;
    int r  = (i >> 4) & 15;
    int hi = (i >> 8) & 1;
    int t  = i >> 9;
    int c  = t * 16 + r;
    int j  = (h >> 3) * 16 + hi * 8 + (h & 7);
    int k  = c >> 3;
    int l  = c & 7;
    Bw[i] = (_Float16)w[(j * DIMS + k) * NL + l];
  }
  for (int j = tid; j < NJ; j += blockDim.x) {
    float acc = 0.f;
    for (int k = 0; k < DIMS; ++k) {
      float m = mu[j * DIMS + k];
      acc += m * m;
    }
    munorm[j] = acc;
    acoef[j]  = -LOG2E / fabsf(s[j]);
  }
}

// ---------------------------------------------------------------------------
// Fused kernel: one wave processes 16 points per tile.
//   2 x v_wmma_f32_16x16x32_f16 : dot[m][j]   = pos . mu_j      (K=16 padded)
//   8 x v_wmma_f32_16x16x32_f16 : mixed[m][c] = gw @ wflat      (K=32, c=k*8+l)
// Then per-lane contraction out[n][l] = sum_k pos[n][k]*mixed[n][k*8+l]
// using the C-fragment ownership pattern (lane&15 -> column, +shfl_xor(8)).
// ---------------------------------------------------------------------------
__launch_bounds__(256, 2)
__global__ void fused_kernel(const float* __restrict__ pos,
                             const _Float16* __restrict__ Bmu,
                             const _Float16* __restrict__ Bw,
                             const float* __restrict__ munorm,
                             const float* __restrict__ acoef,
                             float* __restrict__ out,
                             int nwavesTotal) {
  __shared__ __align__(16) float    sPos[8][16][17];  // f32 pos tile (padded)
  __shared__ __align__(16) float    sPn[8][16];       // ||pos||^2 per row
  __shared__ __align__(16) _Float16 sGw[8][16][40];   // gw tile, padded rows

  const int lane = threadIdx.x & 31;
  const int wv   = threadIdx.x >> 5;
  const int r    = lane & 15;
  const int hi   = lane >> 4;
  const int kbase = hi * 8;

  // ---- resident B fragments (registers) ----
  v16h bmu0 = {}, bmu1 = {};
  {  // K = kbase..kbase+7 in elements 0..7; K 16..31 are zero padding (elements 8..15)
    v8h m0 = *(const v8h*)(Bmu + (0 * 16 + r) * 16 + kbase);
    v8h m1 = *(const v8h*)(Bmu + (1 * 16 + r) * 16 + kbase);
#pragma unroll
    for (int i = 0; i < 8; ++i) { bmu0[i] = m0[i]; bmu1[i] = m1[i]; }
  }
  v16h bw[8];
#pragma unroll
  for (int t = 0; t < 8; ++t)
    bw[t] = *(const v16h*)(Bw + ((t * 2 + hi) * 16 + r) * 16);

  const float mn0 = munorm[r];
  const float mn1 = munorm[16 + r];
  const float a0  = acoef[r];
  const float a1  = acoef[16 + r];

  const int gwave = blockIdx.x * 8 + wv;
  for (int tile = gwave; tile < NTILES; tile += nwavesTotal) {
    const int n0 = tile * 16;

    // prefetch next tile's pos rows into cache
    const int nxt = tile + nwavesTotal;
    if (nxt < NTILES)
      __builtin_prefetch(pos + (size_t)(nxt * 16 + r) * DIMS + kbase, 0, 1);

    // ---- load pos: f16 A-fragment + f32 LDS copy + ||p||^2 ----
    const float* prow = pos + (size_t)(n0 + r) * DIMS + kbase;
    v4f p0 = *(const v4f*)(prow);
    v4f p1 = *(const v4f*)(prow + 4);
    v16h apos = {};
    float sq = 0.f;
#pragma unroll
    for (int i = 0; i < 4; ++i) {
      apos[i]     = (_Float16)p0[i];
      apos[4 + i] = (_Float16)p1[i];
      sPos[wv][r][kbase + i]     = p0[i];
      sPos[wv][r][kbase + 4 + i] = p1[i];
      sq += p0[i] * p0[i] + p1[i] * p1[i];
    }
    sq += __shfl_xor(sq, 16, 32);   // combine K 0..7 with K 8..15
    if (hi == 0) sPn[wv][r] = sq;

    // ---- WMMA 1: dot[m][j] = pos . mu_j ----
    v8f dA = {}, dB = {};
    dA = __builtin_amdgcn_wmma_f32_16x16x32_f16(false, apos, false, bmu0,
                                                (short)0, dA, false, false);
    dB = __builtin_amdgcn_wmma_f32_16x16x32_f16(false, apos, false, bmu1,
                                                (short)0, dB, false, false);

    // ---- gaussian weights: exp2 + row-normalize across 16-lane groups ----
#pragma unroll
    for (int v = 0; v < 8; ++v) {
      const int m = v + 8 * hi;
      const float pn = sPn[wv][m];
      float e0 = EXP2F(a0 * (pn - 2.f * dA[v] + mn0));
      float e1 = EXP2F(a1 * (pn - 2.f * dB[v] + mn1));
      float rs = e0 + e1;
      rs += __shfl_xor(rs, 1, 32);
      rs += __shfl_xor(rs, 2, 32);
      rs += __shfl_xor(rs, 4, 32);
      rs += __shfl_xor(rs, 8, 32);
      const float inv = RCPF(rs + 1e-7f);
      sGw[wv][m][r]      = (_Float16)(e0 * inv);
      sGw[wv][m][16 + r] = (_Float16)(e1 * inv);
    }

    // ---- re-layout gw: C-fragment -> A-fragment via LDS ----
    v8h g0 = *(const v8h*)(&sGw[wv][r][kbase]);       // j = kbase..kbase+7
    v8h g1 = *(const v8h*)(&sGw[wv][r][16 + kbase]);  // j = 16+kbase..
    v16h agw;
#pragma unroll
    for (int i = 0; i < 8; ++i) {
      agw[i]     = g0[i];
      agw[8 + i] = g1[i];
    }

    // ---- WMMA 2: mixed[m][c] = gw[16x32] @ wflat[32x128] ----
    v8f macc[8];
#pragma unroll
    for (int t = 0; t < 8; ++t) {
      macc[t] = (v8f){};
      macc[t] = __builtin_amdgcn_wmma_f32_16x16x32_f16(false, agw, false, bw[t],
                                                       (short)0, macc[t],
                                                       false, false);
    }

    // ---- final contraction: out[n][l] = sum_k pos[n][k]*mixed[n][k*8+l] ----
    const int kodd = r >> 3;  // lane r<8 -> even k, r>=8 -> odd k
    const int l    = r & 7;
#pragma unroll
    for (int v = 0; v < 8; ++v) {
      const int m = v + 8 * hi;
      float acc = 0.f;
#pragma unroll
      for (int t = 0; t < 8; ++t)
        acc += sPos[wv][m][2 * t + kodd] * macc[t][v];
      acc += __shfl_xor(acc, 8, 32);  // even-k + odd-k partials
      if ((lane & 8) == 0)
        out[(size_t)(n0 + m) * NL + l] = acc;
    }
  }
}

extern "C" void kernel_launch(void* const* d_in, const int* in_sizes, int n_in,
                              void* d_out, int out_size, void* d_ws, size_t ws_size,
                              hipStream_t stream) {
  const float* pos = (const float*)d_in[0];  // [N,16]
  const float* mu  = (const float*)d_in[1];  // [32,16]
  const float* s   = (const float*)d_in[2];  // [32]
  const float* w   = (const float*)d_in[3];  // [32,16,8]
  float* out = (float*)d_out;                // [N,8]

  // workspace layout (bytes): Bmu 1024 | Bw 8192 | munorm 128 | acoef 128
  char* ws = (char*)d_ws;
  _Float16* Bmu    = (_Float16*)(ws);
  _Float16* Bw     = (_Float16*)(ws + 1024);
  float*    munorm = (float*)(ws + 1024 + 8192);
  float*    acoef  = (float*)(ws + 1024 + 8192 + 128);

  prep_kernel<<<1, 256, 0, stream>>>(mu, s, w, Bmu, Bw, munorm, acoef);

  const int blocks = 2048;                 // 16384 waves, ~4 tiles each
  const int nwavesTotal = blocks * 8;
  fused_kernel<<<blocks, 256, 0, stream>>>(pos, Bmu, Bw, munorm, acoef, out,
                                           nwavesTotal);
}
